// DecoderLayer_90005334655069
// MI455X (gfx1250) — matrix operand order repaired and checked by reading, not compiled, not run on hardware
//
#include <hip/hip_runtime.h>
#include <hip/hip_bf16.h>
#include <cstdint>
#include <cstddef>

#define BB 4
#define NN 4096
#define EE 128
#define FFD 512

typedef __attribute__((ext_vector_type(16))) _Float16 v16h;
typedef __attribute__((ext_vector_type(8)))  float    v8f;
typedef __attribute__((ext_vector_type(4)))  unsigned int v4u;
typedef __attribute__((ext_vector_type(2)))  unsigned int v2u;
typedef __attribute__((ext_vector_type(4)))  unsigned int u32x4;
typedef __attribute__((ext_vector_type(8)))  int          i32x8;
typedef __attribute__((ext_vector_type(4)))  int          i32x4;

union Frag { v16h v; v4u q[2]; _Float16 h[16]; };
union H4   { _Float16 h[4]; v2u u; };

#if defined(__AMDGCN__) && __has_builtin(__builtin_amdgcn_tensor_load_to_lds) && \
    __has_builtin(__builtin_amdgcn_s_wait_tensorcnt)
#define HAVE_TDM 1
#else
#define HAVE_TDM 0
#endif

__device__ inline v8f vzero() {
  v8f z = {0.f,0.f,0.f,0.f,0.f,0.f,0.f,0.f};
  return z;
}
__device__ inline v8f wmma16(const Frag& a, const Frag& b, v8f c) {
  return __builtin_amdgcn_wmma_f32_16x16x32_f16(false, a.v, false, b.v,
                                                (short)0, c, false, false);
}
__device__ inline unsigned pk2h(float a, float b) {
  union { _Float16 h[2]; unsigned u; } p;
  p.h[0] = (_Float16)a; p.h[1] = (_Float16)b;
  return p.u;
}

#if HAVE_TDM
// TDM: 2-D tile load, dis tensor rows stride NN f32, tile = 64 (contig) x 32 rows.
__device__ __forceinline__ void tdm_load_tile(const float* gsrc, void* ldsdst) {
  unsigned long long ga = (unsigned long long)(const void*)gsrc;
  unsigned lds = (unsigned)(size_t)ldsdst;
  u32x4 g0;
  g0[0] = 1u;                                   // count=1, is_restore=0, gather off
  g0[1] = lds;                                  // lds_addr (bytes)
  g0[2] = (unsigned)ga;                         // global_addr[31:0]
  g0[3] = (unsigned)((ga >> 32) & 0x1FFFFFFu)   // global_addr[56:32]
          | (2u << 30);                         // type=2 (image)
  i32x8 g1;
  g1[0] = (int)(2u << 16);                      // workgroup_mask=0, data_size=4B
  g1[1] = (int)((NN & 0xFFFFu) << 16);          // tensor_dim0[15:0]
  g1[2] = (int)(((NN >> 16) & 0xFFFFu) | ((NN & 0xFFFFu) << 16)); // dim0 hi | dim1 lo
  g1[3] = (int)(((NN >> 16) & 0xFFFFu) | (64u << 16));            // dim1 hi | tile_dim0=64
  g1[4] = 32;                                   // tile_dim1=32, tile_dim2=0
  g1[5] = NN;                                   // tensor_dim0_stride[31:0]
  g1[6] = 0;                                    // stride hi | dim1_stride lo
  g1[7] = 0;
  i32x4 g2 = {0,0,0,0}, g3 = {0,0,0,0};
#if __has_include(<hip/amd_detail/amd_gfx1250_TDM.h>)
  i32x8 g4 = {0,0,0,0,0,0,0,0};
  __builtin_amdgcn_tensor_load_to_lds(g0, g1, g2, g3, g4, 0);
#else
  __builtin_amdgcn_tensor_load_to_lds(g0, g1, g2, g3, 0);
#endif
}
#endif

// ---------------- weight convert + transpose: dst[n][k] = (f16)src[k][n] ----
__global__ void k_cvtT(const float* __restrict__ src, _Float16* __restrict__ dst,
                       int K, int N) {
  int i = blockIdx.x * 256 + threadIdx.x;
  if (i < K * N) {
    int k = i / N, n = i - k * N;
    dst[(size_t)n * K + k] = (_Float16)src[i];
  }
}

// ---------------- fused rmsnorm + QKV projections ---------------------------
// sigqT[b][e][n] = sigmoid(q);  BmT[b][h*32+d][n]=kw*v;  BmT[b][h*32+16+d][n]=kw
__global__ __launch_bounds__(256) void k_qkv(
    const float* __restrict__ x, const float* __restrict__ g_in,
    const _Float16* __restrict__ W16,  // WqT, WkT, WvT each [128][128]
    float* __restrict__ sigqT, _Float16* __restrict__ BmT) {
  __shared__ float shg[EE];
  int tid = threadIdx.x;
  if (tid < EE) shg[tid] = g_in[tid];
  __syncthreads();
  int wave = tid >> 5, lane = tid & 31;
  int t0 = blockIdx.x * 128 + wave * 16;  // 16 tokens per wave, same batch
  int b = t0 >> 12, n0 = t0 & (NN - 1);

  // per-row rsqrt(mean(x^2)+eps): lane pair per row
  float rs;
  {
    int row = lane >> 1;
    const float* xp = x + (size_t)(t0 + row) * EE + (lane & 1) * 64;
    float ss = 0.f;
#pragma unroll
    for (int i = 0; i < 16; i++) {
      float4 v = ((const float4*)xp)[i];
      ss += v.x * v.x + v.y * v.y + v.z * v.z + v.w * v.w;
    }
    ss += __shfl_xor(ss, 1, 32);
    float sc = rsqrtf(ss * (1.0f / EE) + 1e-6f);
    rs = __shfl(sc, (lane & 15) * 2, 32);
  }

  // build A fragments (normalized f16 rows)
  int arow = lane & 15, koff = (lane >> 4) * 8;
  const float* xr = x + (size_t)(t0 + arow) * EE;
  Frag a[4];
#pragma unroll
  for (int kc = 0; kc < 4; kc++) {
    int c0 = kc * 32 + koff;
#pragma unroll
    for (int j = 0; j < 8; j++) {
      a[kc].h[j]     = (_Float16)(xr[c0 + j]      * rs * shg[c0 + j]);
      a[kc].h[j + 8] = (_Float16)(xr[c0 + 16 + j] * rs * shg[c0 + 16 + j]);
    }
  }

  const _Float16* WqT = W16;
  const _Float16* WkT = W16 + EE * EE;
  const _Float16* WvT = W16 + 2 * EE * EE;
#pragma unroll 1
  for (int nt = 0; nt < 8; nt++) {
    v8f aq = vzero(), ak = vzero(), av = vzero();
    int nrow = nt * 16 + (lane & 15);
#pragma unroll
    for (int kc = 0; kc < 4; kc++) {
      size_t off = (size_t)nrow * EE + kc * 32 + (lane >> 4) * 8;
      Frag bq, bk, bv;
      bq.q[0] = *(const v4u*)(WqT + off); bq.q[1] = *(const v4u*)(WqT + off + 16);
      bk.q[0] = *(const v4u*)(WkT + off); bk.q[1] = *(const v4u*)(WkT + off + 16);
      bv.q[0] = *(const v4u*)(WvT + off); bv.q[1] = *(const v4u*)(WvT + off + 16);
      aq = wmma16(a[kc], bq, aq);
      ak = wmma16(a[kc], bk, ak);
      av = wmma16(a[kc], bv, av);
    }
    // epilogue: lane holds col e fixed, 8 consecutive tokens
    int ecol = nt * 16 + (lane & 15);
    int mB = (lane >> 4) * 8;
    float* qp = sigqT + ((size_t)(b * EE + ecol) * NN + n0 + mB);
    float4 q0, q1;
    q0.x = 1.f / (1.f + __expf(-aq[0])); q0.y = 1.f / (1.f + __expf(-aq[1]));
    q0.z = 1.f / (1.f + __expf(-aq[2])); q0.w = 1.f / (1.f + __expf(-aq[3]));
    q1.x = 1.f / (1.f + __expf(-aq[4])); q1.y = 1.f / (1.f + __expf(-aq[5]));
    q1.z = 1.f / (1.f + __expf(-aq[6])); q1.w = 1.f / (1.f + __expf(-aq[7]));
    *(float4*)qp = q0; *(float4*)(qp + 4) = q1;

    Frag fk, fv;
#pragma unroll
    for (int r = 0; r < 8; r++) {
      float kw = __expf(ak[r]);
      fk.h[r] = (_Float16)kw;
      fv.h[r] = (_Float16)(kw * av[r]);
    }
    _Float16* pk = BmT + ((size_t)(b * 256 + nt * 32 + 16 + (lane & 15)) * NN + n0 + mB);
    _Float16* pv = BmT + ((size_t)(b * 256 + nt * 32 + (lane & 15)) * NN + n0 + mB);
    *(v4u*)pk = fk.q[0];
    *(v4u*)pv = fv.q[0];
  }
}

// ---------------- big einsum: outT[b][c][j] = sum_i BmT[b][c][i]*exp(coef*dis[b][i][j])
// A (BmT) streamed from L2; B tile = exp(dis) staged via TDM -> LDS -> f16^T LDS.
__global__ __launch_bounds__(256) void k_attgemm(
    const float* __restrict__ dis, const float* __restrict__ alphap,
    const _Float16* __restrict__ BmT, float* __restrict__ outT) {
#if HAVE_TDM
  __shared__ float raw[2][32][64];      // 16 KB raw f32 tiles (TDM dest)
#endif
  __shared__ _Float16 Bsh[2][64][40];   // transposed f16 tile [j][i], padded
  int tid = threadIdx.x, wave = tid >> 5, lane = tid & 31;
  int b = blockIdx.y;
  int j0 = blockIdx.x * 64;
  float coef = -alphap[0] * 12.0f;      // alpha * log2(4096)

  const float* dbase = dis + (size_t)b * NN * NN + j0;
  // stash mapping: each thread owns an i-pair x j-quad of the 32x64 tile
  int si = (tid >> 4) * 2;   // 0,2,..,30
  int sj = (tid & 15) * 4;   // 0,4,..,60

  v8f acc[2][4];
#pragma unroll
  for (int m = 0; m < 2; m++)
#pragma unroll
    for (int j = 0; j < 4; j++) acc[m][j] = vzero();

  const _Float16* ar0 = BmT + (size_t)(b * 256 + wave * 32 + (lane & 15)) * NN;
  const _Float16* ar1 = ar0 + (size_t)16 * NN;
  int aoff = (lane >> 4) * 8;

#if HAVE_TDM
  if (wave == 0) tdm_load_tile(dbase, &raw[0][0][0]);
#else
  float4 c0, c1;
  {
    const float* p = dbase + (size_t)si * NN + sj;
    c0 = *(const float4*)p; c1 = *(const float4*)(p + NN);
  }
#endif

#pragma unroll 1
  for (int s = 0; s < 128; ++s) {
    int buf = s & 1;
    float4 r0, r1;
#if HAVE_TDM
    if (wave == 0) __builtin_amdgcn_s_wait_tensorcnt(0);  // tile s landed
    __syncthreads();
    if (wave == 0 && s < 127)
      tdm_load_tile(dbase + (size_t)(s + 1) * 32 * NN, &raw[(s + 1) & 1][0][0]);
    r0 = *(const float4*)&raw[buf][si][sj];
    r1 = *(const float4*)&raw[buf][si + 1][sj];
#else
    r0 = c0; r1 = c1;
    if (s < 127) {
      const float* p = dbase + ((size_t)(s + 1) * 32 + si) * NN + sj;
      c0 = *(const float4*)p; c1 = *(const float4*)(p + NN);
    }
    if (s + 4 < 128)
      __builtin_prefetch(dbase + ((size_t)(s + 4) * 32 + si) * NN + sj, 0, 0);
#endif
    {  // exp + pack i-pairs, transposed store (aligned u32 writes)
      float a0[4] = {r0.x, r0.y, r0.z, r0.w};
      float a1[4] = {r1.x, r1.y, r1.z, r1.w};
#pragma unroll
      for (int c = 0; c < 4; c++) {
        unsigned p2 = pk2h(__expf(coef * a0[c]), __expf(coef * a1[c]));
        *(unsigned*)&Bsh[buf][sj + c][si] = p2;
      }
    }
    __syncthreads();

    size_t i0 = (size_t)s * 32;
    Frag fa0, fa1;
    fa0.q[0] = *(const v4u*)(ar0 + i0 + aoff);
    fa0.q[1] = *(const v4u*)(ar0 + i0 + 16 + aoff);
    fa1.q[0] = *(const v4u*)(ar1 + i0 + aoff);
    fa1.q[1] = *(const v4u*)(ar1 + i0 + 16 + aoff);
#pragma unroll
    for (int jt = 0; jt < 4; jt++) {
      Frag bf;
      const _Float16* bp = &Bsh[buf][jt * 16 + (lane & 15)][(lane >> 4) * 8];
      bf.q[0] = *(const v4u*)bp; bf.q[1] = *(const v4u*)(bp + 16);
      acc[0][jt] = wmma16(fa0, bf, acc[0][jt]);
      acc[1][jt] = wmma16(fa1, bf, acc[1][jt]);
    }
    __syncthreads();
  }
#pragma unroll
  for (int mt = 0; mt < 2; mt++) {
    int cb = wave * 32 + mt * 16 + 8 * (lane >> 4);
#pragma unroll
    for (int jt = 0; jt < 4; jt++) {
      int j = j0 + jt * 16 + (lane & 15);
      float* op = outT + (size_t)(b * 256 + cb) * NN + j;
#pragma unroll
      for (int r = 0; r < 8; r++) op[(size_t)r * NN] = acc[mt][jt][r];
    }
  }
}

// ---------------- att = sigmoid(q)*w1/w2, transpose to [b][n][e] f16 --------
__global__ __launch_bounds__(256) void k_att(
    const float* __restrict__ sigqT, const float* __restrict__ outT,
    _Float16* __restrict__ att16) {
  __shared__ _Float16 sh[EE][34];
  int b = blockIdx.y, n0 = blockIdx.x * 32;
  int tid = threadIdx.x;
  int e = tid >> 1, nc = (tid & 1) * 16;
  int hh = e >> 4, dd = e & 15;
  const float* p1 = outT + (size_t)(b * 256 + hh * 32 + dd) * NN + n0 + nc;
  const float* p2 = outT + (size_t)(b * 256 + hh * 32 + 16 + dd) * NN + n0 + nc;
  const float* ps = sigqT + (size_t)(b * EE + e) * NN + n0 + nc;
#pragma unroll
  for (int i = 0; i < 16; i++)
    sh[e][nc + i] = (_Float16)(ps[i] * p1[i] / p2[i]);
  __syncthreads();
  int n = tid >> 3, eb = (tid & 7) * 16;
  Frag f;
#pragma unroll
  for (int i = 0; i < 16; i++) f.h[i] = sh[eb + i][n];
  _Float16* op = att16 + ((size_t)(b * NN + n0 + n) * EE + eb);
  *(v4u*)op = f.q[0]; *(v4u*)(op + 8) = f.q[1];
}

// ---------------- out1 = x + att @ Wo ---------------------------------------
__global__ __launch_bounds__(256) void k_wo(
    const _Float16* __restrict__ att16, const _Float16* __restrict__ WoT,
    const float* __restrict__ x, float* __restrict__ out) {
  int tid = threadIdx.x, wave = tid >> 5, lane = tid & 31;
  int t0 = blockIdx.x * 128 + wave * 16;
  int koff = (lane >> 4) * 8;
  const _Float16* ap = att16 + (size_t)(t0 + (lane & 15)) * EE;
  Frag a[4];
#pragma unroll
  for (int kc = 0; kc < 4; kc++) {
    a[kc].q[0] = *(const v4u*)(ap + kc * 32 + koff);
    a[kc].q[1] = *(const v4u*)(ap + kc * 32 + 16 + koff);
  }
#pragma unroll 1
  for (int nt = 0; nt < 8; nt++) {
    v8f acc = vzero();
    int nrow = nt * 16 + (lane & 15);
#pragma unroll
    for (int kc = 0; kc < 4; kc++) {
      size_t off = (size_t)nrow * EE + kc * 32 + (lane >> 4) * 8;
      Frag bf;
      bf.q[0] = *(const v4u*)(WoT + off); bf.q[1] = *(const v4u*)(WoT + off + 16);
      acc = wmma16(a[kc], bf, acc);
    }
    int ecol = nt * 16 + (lane & 15), mB = 8 * (lane >> 4);
#pragma unroll
    for (int r = 0; r < 8; r++) {
      size_t idx = (size_t)(t0 + mB + r) * EE + ecol;
      out[idx] = x[idx] + acc[r];
    }
  }
}

// ---------------- h2 = rmsnorm(out1, g_post) f16 ----------------------------
__global__ __launch_bounds__(256) void k_rms2(const float* __restrict__ out1,
                                              const float* __restrict__ g_post,
                                              _Float16* __restrict__ h2) {
  int tid = threadIdx.x, wave = tid >> 5, lane = tid & 31;
  int t = blockIdx.x * 8 + wave;
  float4 v = *(const float4*)(out1 + (size_t)t * EE + lane * 4);
  float ss = v.x * v.x + v.y * v.y + v.z * v.z + v.w * v.w;
#pragma unroll
  for (int o = 1; o < 32; o <<= 1) ss += __shfl_xor(ss, o, 32);
  float sc = rsqrtf(ss * (1.0f / EE) + 1e-6f);
  float4 g = *(const float4*)(g_post + lane * 4);
  H4 p;
  p.h[0] = (_Float16)(v.x * sc * g.x); p.h[1] = (_Float16)(v.y * sc * g.y);
  p.h[2] = (_Float16)(v.z * sc * g.z); p.h[3] = (_Float16)(v.w * sc * g.w);
  *(v2u*)(h2 + (size_t)t * EE + lane * 4) = p.u;
}

// ---------------- ff1 = relu(h2 @ W1 + b1) f16 ------------------------------
__global__ __launch_bounds__(256) void k_ffn1(
    const _Float16* __restrict__ h2, const _Float16* __restrict__ W1T,
    const float* __restrict__ b1, _Float16* __restrict__ ff1) {
  int tid = threadIdx.x, wave = tid >> 5, lane = tid & 31;
  int t0 = blockIdx.x * 128 + wave * 16;
  int koff = (lane >> 4) * 8;
  const _Float16* ap = h2 + (size_t)(t0 + (lane & 15)) * EE;
  Frag a[4];
#pragma unroll
  for (int kc = 0; kc < 4; kc++) {
    a[kc].q[0] = *(const v4u*)(ap + kc * 32 + koff);
    a[kc].q[1] = *(const v4u*)(ap + kc * 32 + 16 + koff);
  }
#pragma unroll 1
  for (int nt = 0; nt < 32; nt++) {
    v8f acc = vzero();
    int nrow = nt * 16 + (lane & 15);
#pragma unroll
    for (int kc = 0; kc < 4; kc++) {
      size_t off = (size_t)nrow * EE + kc * 32 + (lane >> 4) * 8;
      Frag bf;
      bf.q[0] = *(const v4u*)(W1T + off); bf.q[1] = *(const v4u*)(W1T + off + 16);
      acc = wmma16(a[kc], bf, acc);
    }
    int col = nt * 16 + (lane & 15), mB = 8 * (lane >> 4);
    float bias = b1[col];
#pragma unroll
    for (int r = 0; r < 8; r++) {
      float val = fmaxf(acc[r] + bias, 0.f);
      ff1[(size_t)(t0 + mB + r) * FFD + col] = (_Float16)val;
    }
  }
}

// ---------------- out += ff1 @ W2 + b2 --------------------------------------
__global__ __launch_bounds__(256) void k_ffn2(
    const _Float16* __restrict__ ff1, const _Float16* __restrict__ W2T,
    const float* __restrict__ b2, float* __restrict__ out) {
  int tid = threadIdx.x, wave = tid >> 5, lane = tid & 31;
  int t0 = blockIdx.x * 128 + wave * 16;
  int koff = (lane >> 4) * 8;
  const _Float16* ap = ff1 + (size_t)(t0 + (lane & 15)) * FFD;
  v8f acc[8];
#pragma unroll
  for (int i = 0; i < 8; i++) acc[i] = vzero();
#pragma unroll 1
  for (int kc = 0; kc < 16; kc++) {
    Frag a;
    a.q[0] = *(const v4u*)(ap + kc * 32 + koff);
    a.q[1] = *(const v4u*)(ap + kc * 32 + 16 + koff);
#pragma unroll
    for (int nt = 0; nt < 8; nt++) {
      size_t off = (size_t)(nt * 16 + (lane & 15)) * FFD + kc * 32 + (lane >> 4) * 8;
      Frag bf;
      bf.q[0] = *(const v4u*)(W2T + off); bf.q[1] = *(const v4u*)(W2T + off + 16);
      acc[nt] = wmma16(a, bf, acc[nt]);
    }
  }
#pragma unroll
  for (int nt = 0; nt < 8; nt++) {
    int ecol = nt * 16 + (lane & 15), mB = 8 * (lane >> 4);
    float bias = b2[ecol];
#pragma unroll
    for (int r = 0; r < 8; r++) {
      size_t idx = (size_t)(t0 + mB + r) * EE + ecol;
      out[idx] += acc[nt][r] + bias;
    }
  }
}

// ---------------------------------------------------------------------------
extern "C" void kernel_launch(void* const* d_in, const int* in_sizes, int n_in,
                              void* d_out, int out_size, void* d_ws, size_t ws_size,
                              hipStream_t stream) {
  const float* x      = (const float*)d_in[0];
  const float* dis    = (const float*)d_in[1];
  const float* g_in   = (const float*)d_in[2];
  const float* g_post = (const float*)d_in[3];
  const float* Wq     = (const float*)d_in[4];
  const float* Wk     = (const float*)d_in[5];
  const float* Wv     = (const float*)d_in[6];
  const float* alpha  = (const float*)d_in[7];
  const float* Wo     = (const float*)d_in[8];
  const float* W1     = (const float*)d_in[9];
  const float* b1     = (const float*)d_in[10];
  const float* W2     = (const float*)d_in[11];
  const float* b2     = (const float*)d_in[12];
  float* out = (float*)d_out;

  char* ws = (char*)d_ws;
  const size_t MB = 1024ull * 1024ull;
  _Float16* W16   = (_Float16*)(ws);                  // 384 KB
  float*    sigqT = (float*)(ws + 512 * 1024);        // 8 MB f32 [4][128][4096]
  _Float16* BmT   = (_Float16*)(ws + 512 * 1024 + 8 * MB);   // 8 MB f16 [4][256][4096]
  float*    outT  = (float*)(ws + 512 * 1024 + 16 * MB);     // 16 MB f32 [4][256][4096]
  _Float16* att16 = (_Float16*)BmT;    // alias: BmT dead after k_attgemm
  _Float16* h2    = (_Float16*)sigqT;  // alias: sigqT dead after k_att
  _Float16* ff1   = (_Float16*)outT;   // alias: outT dead after k_att

  // weight convert + transpose to f16 [n][k]
  k_cvtT<<<(EE * EE + 255) / 256, 256, 0, stream>>>(Wq, W16 + 0,      EE, EE);
  k_cvtT<<<(EE * EE + 255) / 256, 256, 0, stream>>>(Wk, W16 + 16384,  EE, EE);
  k_cvtT<<<(EE * EE + 255) / 256, 256, 0, stream>>>(Wv, W16 + 32768,  EE, EE);
  k_cvtT<<<(EE * EE + 255) / 256, 256, 0, stream>>>(Wo, W16 + 49152,  EE, EE);
  k_cvtT<<<(EE * FFD + 255) / 256, 256, 0, stream>>>(W1, W16 + 65536, EE, FFD);
  k_cvtT<<<(FFD * EE + 255) / 256, 256, 0, stream>>>(W2, W16 + 131072, FFD, EE);

  k_qkv<<<128, 256, 0, stream>>>(x, g_in, W16, sigqT, BmT);
  dim3 g2(64, 4);
  k_attgemm<<<g2, 256, 0, stream>>>(dis, alpha, BmT, outT);
  dim3 g3(128, 4);
  k_att<<<g3, 256, 0, stream>>>(sigqT, outT, att16);
  k_wo<<<128, 256, 0, stream>>>(att16, W16 + 49152, x, out);
  k_rms2<<<2048, 256, 0, stream>>>(out, g_post, h2);
  k_ffn1<<<128, 256, 0, stream>>>(h2, W16 + 65536, b1, ff1);
  k_ffn2<<<128, 256, 0, stream>>>(ff1, W16 + 131072, b2, out);

  (void)in_sizes; (void)n_in; (void)out_size; (void)ws_size;
}
